// AttentionBase_7653631721597
// MI455X (gfx1250) — matrix operand order repaired and checked by reading, not compiled, not run on hardware
//
#include <hip/hip_runtime.h>
#include <hip/hip_bf16.h>
#include <cstdint>
#include <cstddef>

// Problem constants (match reference)
#define TT  2048
#define DD  2048
#define NH  16
#define HD  128
#define WIN 512

typedef _Float16 half_t;
typedef __attribute__((ext_vector_type(4)))  _Float16 v4h;
typedef __attribute__((ext_vector_type(8)))  _Float16 v8h;
typedef __attribute__((ext_vector_type(16))) _Float16 v16h;
typedef __attribute__((ext_vector_type(8)))  float    v8f;
typedef __attribute__((ext_vector_type(4)))  unsigned int u32x4;
typedef __attribute__((ext_vector_type(8)))  int      i32x8;
typedef __attribute__((ext_vector_type(4)))  int      i32x4;

#ifndef __has_builtin
#define __has_builtin(x) 0
#endif

#if __has_builtin(__builtin_amdgcn_tensor_load_to_lds)
#define HAVE_TDM 1
#else
#define HAVE_TDM 0
#endif

// ---------------------------------------------------------------------------
// TDM helpers (CDNA5 Tensor Data Mover, ISA ch. 8 descriptor layout)
// ---------------------------------------------------------------------------
__device__ __forceinline__ void wait_tensor0() {
#if __has_builtin(__builtin_amdgcn_s_wait_tensorcnt)
  __builtin_amdgcn_s_wait_tensorcnt(0);
#else
  asm volatile("s_wait_tensorcnt 0x0" ::: "memory");
#endif
}

__device__ __forceinline__ uint32_t lds_off_u32(const void* p) {
  // generic LDS pointer: addr[31:0] is the LDS byte offset (ISA 10.2 aperture map)
  return (uint32_t)(uintptr_t)p;
}

#if HAVE_TDM
// 2-D tile load: tile_dim0 = 64 elems (contiguous, 2B each), tile_dim1 = 128 rows,
// row stride = row_stride_elems. LDS receives a dense 128x64 f16 tile.
__device__ __forceinline__ void tdm_load_tile_2d(const half_t* gsrc,
                                                 int row_stride_elems,
                                                 uint32_t lds_byte_addr) {
  const uint64_t ga = (uint64_t)(uintptr_t)gsrc;
  u32x4 g0;
  g0.x = 1u;                                                  // count=1 (valid user D#)
  g0.y = lds_byte_addr;                                       // lds_addr
  g0.z = (uint32_t)ga;                                        // global_addr[31:0]
  g0.w = (uint32_t)((ga >> 32) & 0x01FFFFFFu) | (2u << 30);   // addr[56:32] | type=2
  i32x8 g1;
  g1[0] = 1 << 16;                          // workgroup_mask=0, data_size=1 (2 bytes)
  g1[1] = (int)(2048u << 16);               // tensor_dim0 low16 -> bits[79:64+? ] ([31:16])
  g1[2] = (int)(2048u << 16);               // dim0 hi16=0 | tensor_dim1 low16
  g1[3] = 64 << 16;                         // dim1 hi16=0 | tile_dim0 = 64
  g1[4] = 128;                              // tile_dim1 = 128, tile_dim2 = 0
  g1[5] = row_stride_elems;                 // tensor_dim0_stride[31:0]
  g1[6] = 0;                                // stride0 hi | dim1_stride low16
  g1[7] = 0;
  const i32x4 z4 = {0, 0, 0, 0};
#if defined(__clang_major__) && __clang_major__ >= 23
  const i32x8 z8 = {0, 0, 0, 0, 0, 0, 0, 0};
  __builtin_amdgcn_tensor_load_to_lds(g0, g1, z4, z4, z8, 0);
#else
  __builtin_amdgcn_tensor_load_to_lds(g0, g1, z4, z4, 0);
#endif
}
#endif  // HAVE_TDM

// ---------------------------------------------------------------------------
// WMMA fragment loaders (wave32; A/B in contiguous-k order chosen by this code, C/D per ISA)
// A (16x32 f16): lane L holds row M = L&15. halves 0..15 -> K = base+lg*16+0..15
//                (lg = L>>4)
// B (32x16 f16): lane L holds col N = L&15. halves 0..15 -> K = base+lg*16+0..15
// C/D (16x16 f32): lane L col N=L&15, vgpr r -> row M = r + lg*8
// ---------------------------------------------------------------------------
__device__ __forceinline__ v16h load_frag_a(const half_t* p) {
  v8h lo = *(const v8h*)p;
  v8h hi = *(const v8h*)(p + 8);
  return __builtin_shufflevector(lo, hi, 0,1,2,3,4,5,6,7,8,9,10,11,12,13,14,15);
}
__device__ __forceinline__ v16h load_frag_b(const half_t* p) {
  v8h lo = *(const v8h*)p;
  v8h hi = *(const v8h*)(p + 8);
  return __builtin_shufflevector(lo, hi, 0,1,2,3,4,5,6,7,8,9,10,11,12,13,14,15);
}

// ---------------------------------------------------------------------------
// fp32 -> f16 conversion (vectorized, grid-stride)
// ---------------------------------------------------------------------------
__global__ __launch_bounds__(256) void cvt_f16_kernel(const float* __restrict__ src,
                                                      half_t* __restrict__ dst, int n4) {
  int i = blockIdx.x * blockDim.x + threadIdx.x;
  int stride = gridDim.x * blockDim.x;
  for (; i < n4; i += stride) {
    float4 f = ((const float4*)src)[i];
    v4h o = {(half_t)f.x, (half_t)f.y, (half_t)f.z, (half_t)f.w};
    ((v4h*)dst)[i] = o;
  }
}

// ---------------------------------------------------------------------------
// Tiled f16 WMMA GEMM: C[M,N] = A[M,K] * B[K,N]
// Block = 256 threads (8 waves). Tile 128x128, BK=64, double-buffered LDS.
// A tile staged by the Tensor Data Mover (wave 0 issues the descriptor; TDM
// ignores EXEC so gating by wave avoids duplicate DMAs); B tile transposed
// into LDS manually. OutT in {half_t, float}; TRANS stores C^T (ld = M).
// ---------------------------------------------------------------------------
template <typename OutT, bool TRANS>
__global__ __launch_bounds__(256) void gemm_kernel(const half_t* __restrict__ A,
                                                   const half_t* __restrict__ B,
                                                   OutT* __restrict__ C,
                                                   int M, int N, int K) {
  __shared__ half_t As[2][128 * 64];   // [m][k]
  __shared__ half_t Bs[2][128 * 64];   // [n][k] (transposed on store)
  const int tid  = threadIdx.x;
  const int wave = tid >> 5;
  const int lane = tid & 31;
  const int lg   = lane >> 4;
  const int lr   = lane & 15;
  const int m0 = blockIdx.y * 128;
  const int n0 = blockIdx.x * 128;
  const int wm = (wave & 1) * 64;
  const int wn = (wave >> 1) * 32;

  v8f acc[4][2] = {};

  auto stage = [&](int buf, int k0) {
#if HAVE_TDM
    if (tid < 32)
      tdm_load_tile_2d(&A[(size_t)m0 * K + k0], K, lds_off_u32(&As[buf][0]));
#else
#pragma unroll
    for (int i = 0; i < 4; ++i) {
      int c = tid + 256 * i;
      int r = c >> 3, col = (c & 7) * 8;
      *(v8h*)&As[buf][r * 64 + col] = *(const v8h*)&A[(size_t)(m0 + r) * K + k0 + col];
    }
#endif
#pragma unroll
    for (int i = 0; i < 4; ++i) {
      int c = tid + 256 * i;
      int kr = c >> 4, nc = (c & 15) * 8;
      v8h v = *(const v8h*)&B[(size_t)(k0 + kr) * N + n0 + nc];
#pragma unroll
      for (int j = 0; j < 8; ++j) Bs[buf][(size_t)(nc + j) * 64 + kr] = v[j];
    }
  };
  auto wait_stage = [&]() {
#if HAVE_TDM
    if (tid < 32) wait_tensor0();
#endif
  };

  stage(0, 0);
  wait_stage();
  __syncthreads();

  int buf = 0;
  const int nk = K >> 6;
  for (int it = 0; it < nk; ++it) {
    if (it + 1 < nk) stage(buf ^ 1, (it + 1) << 6);

#pragma unroll
    for (int ks = 0; ks < 2; ++ks) {
      v16h a[4], b[2];
#pragma unroll
      for (int mi = 0; mi < 4; ++mi)
        a[mi] = load_frag_a(&As[buf][(wm + mi * 16 + lr) * 64 + ks * 32 + lg * 16]);
#pragma unroll
      for (int ni = 0; ni < 2; ++ni)
        b[ni] = load_frag_b(&Bs[buf][(wn + ni * 16 + lr) * 64 + ks * 32 + lg * 16]);
#pragma unroll
      for (int mi = 0; mi < 4; ++mi)
#pragma unroll
        for (int ni = 0; ni < 2; ++ni)
          acc[mi][ni] = __builtin_amdgcn_wmma_f32_16x16x32_f16(
              false, a[mi], false, b[ni], (short)0, acc[mi][ni], false, false);
    }

    wait_stage();     // wave 0: TDM fill of buf^1 complete
    __syncthreads();  // all waves: done reading buf; B stores visible
    buf ^= 1;
  }

#pragma unroll
  for (int mi = 0; mi < 4; ++mi)
#pragma unroll
    for (int ni = 0; ni < 2; ++ni)
#pragma unroll
      for (int r = 0; r < 8; ++r) {
        int row = m0 + wm + mi * 16 + r + lg * 8;
        int col = n0 + wn + ni * 16 + lr;
        float v = acc[mi][ni][r];
        if (TRANS) C[(size_t)col * M + row] = (OutT)v;
        else       C[(size_t)row * N + col] = (OutT)v;
      }
}

// ---------------------------------------------------------------------------
// Fused per-(t,h)-row RMSNorm (over HD=128) + RoPE, in-place on f16 buffer.
// One wave per row; 4 elements (2 rope pairs) per lane.
// ---------------------------------------------------------------------------
__global__ __launch_bounds__(256) void norm_rope_kernel(half_t* __restrict__ X,
                                                        const float* __restrict__ w,
                                                        const int* __restrict__ pos) {
  const int wave = threadIdx.x >> 5;
  const int lane = threadIdx.x & 31;
  const int row  = blockIdx.x * 8 + wave;   // [0, TT*NH)
  const int t = row >> 4;                   // NH == 16
  const int h = row & 15;
  half_t* p = X + (size_t)t * DD + h * HD + lane * 4;

  v4h xv = *(const v4h*)p;
  float x0 = (float)xv[0], x1 = (float)xv[1], x2 = (float)xv[2], x3 = (float)xv[3];
  float ss = x0 * x0 + x1 * x1 + x2 * x2 + x3 * x3;
#pragma unroll
  for (int off = 16; off > 0; off >>= 1) ss += __shfl_xor(ss, off);
  const float rinv = rsqrtf(ss * (1.0f / 128.0f) + 1e-6f);

  x0 *= rinv * w[lane * 4 + 0];
  x1 *= rinv * w[lane * 4 + 1];
  x2 *= rinv * w[lane * 4 + 2];
  x3 *= rinv * w[lane * 4 + 3];

  const float posf = (float)pos[t];
  const float kfreq = 0.14391156831212787f;   // ln(10000)/64
  const int j0 = lane * 2, j1 = lane * 2 + 1;
  float a0 = posf * __expf(-(float)j0 * kfreq);
  float a1 = posf * __expf(-(float)j1 * kfreq);
  float c0, s0, c1, s1;
  __sincosf(a0, &s0, &c0);
  __sincosf(a1, &s1, &c1);

  v4h o = {(half_t)(x0 * c0 - x1 * s0), (half_t)(x0 * s0 + x1 * c0),
           (half_t)(x2 * c1 - x3 * s1), (half_t)(x2 * s1 + x3 * c1)};
  *(v4h*)p = o;
}

// ---------------------------------------------------------------------------
// Flash attention with sliding window (WIN=512), WMMA f16, one wave per
// (head, 16-row q-tile). Key blocks of 64. Q/K read directly (L2-resident);
// V read from transposed Vt[d][t] so PV B-fragments are contiguous.
// ---------------------------------------------------------------------------
__global__ __launch_bounds__(256) void attn_kernel(const half_t* __restrict__ Q,
                                                   const half_t* __restrict__ K,
                                                   const half_t* __restrict__ Vt,
                                                   half_t* __restrict__ Ctx) {
  __shared__ half_t Ps[8][16 * 64];   // per-wave exp(S) staging, 2KB each
  const int wave = threadIdx.x >> 5;
  const int lane = threadIdx.x & 31;
  const int lg   = lane >> 4;
  const int lr   = lane & 15;
  const int gw = blockIdx.x * 8 + wave;     // [0, 2048)
  const int h  = gw & (NH - 1);
  const int q0 = (gw >> 4) * 16;
  const float scale = 0.088388347648318447f;  // 1/sqrt(128)

  v16h qf[4];
#pragma unroll
  for (int kf = 0; kf < 4; ++kf)
    qf[kf] = load_frag_a(Q + (size_t)(q0 + lr) * DD + h * HD + kf * 32 + lg * 16);

  float m_[8], l_[8];
  v8f o[8] = {};
#pragma unroll
  for (int r = 0; r < 8; ++r) { m_[r] = -1.0e30f; l_[r] = 0.0f; }

  int slo = q0 - (WIN - 1);
  if (slo < 0) slo = 0;
  const int kb0 = slo & ~63;
  const int kb1 = q0 + 15;
  half_t* ps = Ps[wave];

  for (int kb = kb0; kb <= kb1; kb += 64) {
    // ---- S = Q K^T (16 x 64) ----
    v8f s[4] = {};
#pragma unroll
    for (int nf = 0; nf < 4; ++nf) {
      const size_t kbase = (size_t)(kb + nf * 16 + lr) * DD + h * HD;
#pragma unroll
      for (int kf = 0; kf < 4; ++kf) {
        v16h b = load_frag_b(K + kbase + kf * 32 + lg * 16);
        s[nf] = __builtin_amdgcn_wmma_f32_16x16x32_f16(
            false, qf[kf], false, b, (short)0, s[nf], false, false);
      }
    }
    // ---- scale + window mask + row max ----
    float rmax[8];
#pragma unroll
    for (int r = 0; r < 8; ++r) rmax[r] = -3.0e38f;
#pragma unroll
    for (int nf = 0; nf < 4; ++nf) {
      const int key = kb + nf * 16 + lr;
#pragma unroll
      for (int r = 0; r < 8; ++r) {
        const int t = q0 + r + lg * 8;
        float v = s[nf][r] * scale;
        const bool ok = (key <= t) && ((t - key) < WIN);
        v = ok ? v : -3.0e38f;
        s[nf][r] = v;
        rmax[r] = fmaxf(rmax[r], v);
      }
    }
#pragma unroll
    for (int r = 0; r < 8; ++r) {
#pragma unroll
      for (int off = 8; off > 0; off >>= 1)
        rmax[r] = fmaxf(rmax[r], __shfl_xor(rmax[r], off));
    }
    // ---- online softmax update ----
    float alpha[8];
#pragma unroll
    for (int r = 0; r < 8; ++r) {
      float mn = fmaxf(m_[r], rmax[r]);
      alpha[r] = __expf(m_[r] - mn);
      m_[r] = mn;
      l_[r] *= alpha[r];
    }
#pragma unroll
    for (int nf = 0; nf < 8; ++nf)
#pragma unroll
      for (int r = 0; r < 8; ++r) o[nf][r] *= alpha[r];

    float rsum[8];
#pragma unroll
    for (int r = 0; r < 8; ++r) rsum[r] = 0.0f;
#pragma unroll
    for (int nf = 0; nf < 4; ++nf) {
#pragma unroll
      for (int r = 0; r < 8; ++r) {
        float pv = __expf(s[nf][r] - m_[r]);
        rsum[r] += pv;
        ps[(r + lg * 8) * 64 + nf * 16 + lr] = (half_t)pv;   // C->A layout via LDS
      }
    }
#pragma unroll
    for (int r = 0; r < 8; ++r) {
      float tsum = rsum[r];
#pragma unroll
      for (int off = 8; off > 0; off >>= 1) tsum += __shfl_xor(tsum, off);
      l_[r] += tsum;
    }
    // ---- O += P V  (16 x 128) ----
#pragma unroll
    for (int ks = 0; ks < 2; ++ks) {
      v16h pa = load_frag_a(&ps[lr * 64 + ks * 32 + lg * 16]);
#pragma unroll
      for (int nf = 0; nf < 8; ++nf) {
        v16h b = load_frag_b(Vt + (size_t)(h * HD + nf * 16 + lr) * TT +
                             kb + ks * 32 + lg * 16);
        o[nf] = __builtin_amdgcn_wmma_f32_16x16x32_f16(
            false, pa, false, b, (short)0, o[nf], false, false);
      }
    }
  }

  // ---- finalize: O / l, store context as f16 ----
  float inv[8];
#pragma unroll
  for (int r = 0; r < 8; ++r) inv[r] = 1.0f / l_[r];
#pragma unroll
  for (int nf = 0; nf < 8; ++nf)
#pragma unroll
    for (int r = 0; r < 8; ++r) {
      const int t = q0 + r + lg * 8;
      Ctx[(size_t)t * DD + h * HD + nf * 16 + lr] = (half_t)(o[nf][r] * inv[r]);
    }
}

// ---------------------------------------------------------------------------
// Launcher
// d_in: [hidden_states, wq, wk, wv, wo, q_norm_w, k_norm_w, position_ids]
// ---------------------------------------------------------------------------
extern "C" void kernel_launch(void* const* d_in, const int* in_sizes, int n_in,
                              void* d_out, int out_size, void* d_ws, size_t ws_size,
                              hipStream_t stream) {
  const float* X   = (const float*)d_in[0];
  const float* Wq  = (const float*)d_in[1];
  const float* Wk  = (const float*)d_in[2];
  const float* Wv  = (const float*)d_in[3];
  const float* Wo  = (const float*)d_in[4];
  const float* qw  = (const float*)d_in[5];
  const float* kw  = (const float*)d_in[6];
  const int*   pos = (const int*)d_in[7];
  float* out = (float*)d_out;

  uint8_t* ws = (uint8_t*)d_ws;
  const size_t HB = (size_t)TT * DD * sizeof(half_t);   // 8 MiB per buffer
  half_t* Xh  = (half_t*)(ws + 0 * HB);
  half_t* Wqh = (half_t*)(ws + 1 * HB);
  half_t* Wkh = (half_t*)(ws + 2 * HB);
  half_t* Wvh = (half_t*)(ws + 3 * HB);
  half_t* Woh = (half_t*)(ws + 4 * HB);
  half_t* Qh  = (half_t*)(ws + 5 * HB);
  half_t* Kh  = (half_t*)(ws + 6 * HB);
  half_t* Vth = (half_t*)(ws + 7 * HB);   // V transposed: [D][T]
  half_t* Cxh = (half_t*)(ws + 8 * HB);   // context, f16

  const int n4 = (TT * DD) / 4;
  cvt_f16_kernel<<<4096, 256, 0, stream>>>(X,  Xh,  n4);
  cvt_f16_kernel<<<4096, 256, 0, stream>>>(Wq, Wqh, n4);
  cvt_f16_kernel<<<4096, 256, 0, stream>>>(Wk, Wkh, n4);
  cvt_f16_kernel<<<4096, 256, 0, stream>>>(Wv, Wvh, n4);
  cvt_f16_kernel<<<4096, 256, 0, stream>>>(Wo, Woh, n4);

  dim3 gg(TT / 128, TT / 128);
  gemm_kernel<half_t, false><<<gg, 256, 0, stream>>>(Xh, Wqh, Qh,  TT, DD, DD);
  gemm_kernel<half_t, false><<<gg, 256, 0, stream>>>(Xh, Wkh, Kh,  TT, DD, DD);
  gemm_kernel<half_t, true ><<<gg, 256, 0, stream>>>(Xh, Wvh, Vth, TT, DD, DD);

  norm_rope_kernel<<<(TT * NH) / 8, 256, 0, stream>>>(Qh, qw, pos);
  norm_rope_kernel<<<(TT * NH) / 8, 256, 0, stream>>>(Kh, kw, pos);

  attn_kernel<<<(TT / 16) * NH / 8, 256, 0, stream>>>(Qh, Kh, Vth, Cxh);

  gemm_kernel<float, false><<<gg, 256, 0, stream>>>(Cxh, Woh, out, TT, DD, DD);
}
